// SpGraphTransAttentionLayer_34170759807545
// MI455X (gfx1250) — compile-verified, matrix operand order checked by reading, not run on hardware
//
#include <hip/hip_runtime.h>
#include <hip/hip_bf16.h>

// ---------------------------------------------------------------------------
// SpGraphTransAttentionLayer for MI455X (gfx1250, wave32, WMMA)
//   q/k/v = x @ W^T + b  via v_wmma_f32_16x16x32_f16  (f16 in, f32 acc)
//   - A tile (16x256) staged once per block in LDS (shared by all 8 waves)
//   - each wave register-blocks 2 N-tiles (16 WMMA per wave)
//   edge logits + scatter softmax over edge[0] (q/k kept f16, L2-resident)
// Outputs (concat): attention [E,8] f32, v [N,32,8] f32 (permuted store)
// ---------------------------------------------------------------------------

typedef __attribute__((ext_vector_type(16))) _Float16 v16h;
typedef __attribute__((ext_vector_type(8)))  _Float16 v8h;
typedef __attribute__((ext_vector_type(4)))  _Float16 v4h;
typedef __attribute__((ext_vector_type(8)))  float    v8f;

#define ATT_D 256
#define IN_F  256
#define NHEAD 8
#define DK    32
#define LDS_PITCH 264   // 256 + 8 halves pad: 528B row stride -> conflict-free ds_load_b128

// --- helpers ---------------------------------------------------------------

__device__ __forceinline__ v16h fuse16(v8h lo, v8h hi) {
    return __builtin_shufflevector(lo, hi, 0,1,2,3,4,5,6,7,8,9,10,11,12,13,14,15);
}

__device__ __forceinline__ v16h load_half16(const _Float16* p) {
    // lane-local A/B fragment: 8 contiguous halves at p, 8 more at p+16
    return fuse16(*(const v8h*)(p), *(const v8h*)(p + 16));
}

__device__ __forceinline__ void atomicMaxFloat(float* addr, float val) {
    // standard monotone punning trick; mbuf initialized to -inf
    if (!(val < 0.0f)) atomicMax((int*)addr, __float_as_int(val));
    else               atomicMin((unsigned int*)addr, __float_as_uint(val));
}

// --- conversion kernels ----------------------------------------------------

__global__ void __launch_bounds__(256)
cvt_f32_to_f16(const float* __restrict__ in, _Float16* __restrict__ out, int n) {
    int t = (blockIdx.x * blockDim.x + threadIdx.x) * 4;
    if (t + 3 < n) {
        float4 v = *(const float4*)(in + t);
        v4h h; h[0] = (_Float16)v.x; h[1] = (_Float16)v.y;
               h[2] = (_Float16)v.z; h[3] = (_Float16)v.w;
        *(v4h*)(out + t) = h;
    } else {
        for (int i = t; i < n; ++i) out[i] = (_Float16)in[i];
    }
}

__global__ void __launch_bounds__(256)
init_max_sum(float* __restrict__ m, float* __restrict__ s, int n) {
    int t = blockIdx.x * blockDim.x + threadIdx.x;
    if (t < n) { m[t] = -__builtin_inff(); s[t] = 0.0f; }
}

// --- WMMA GEMM: out[M,256] = A[M,256] @ W[256,256]^T + bias ----------------
// One block = one 16-row M tile x all 256 cols. A slab staged in LDS.
// Each wave owns 2 column tiles (2 accumulators, 16 WMMAs).
// MODE 0: store f16 (q/k staging in ws)
// MODE 1: store f32 with [N,H,DK]->[N,DK,H] permutation (v directly to d_out)

template<int MODE>
__global__ void __launch_bounds__(256)
gemm_qkv_wmma(const _Float16* __restrict__ A, const _Float16* __restrict__ W,
              const float* __restrict__ bias, _Float16* __restrict__ outH,
              float* __restrict__ outF, int M)
{
    __shared__ _Float16 As[16 * LDS_PITCH];

    const int lane = threadIdx.x & 31;
    const int wave = threadIdx.x >> 5;
    const int tM   = blockIdx.x;

    // --- cooperative stage of A slab: 16 rows x 256 halves (8 KB) ----------
    {
        const int r   = threadIdx.x >> 4;          // 0..15
        const int cs  = (threadIdx.x & 15) << 4;   // 16-half segment
        int grow = tM * 16 + r;
        if (grow >= M) grow = M - 1;               // clamp (stores guarded)
        const _Float16* gp = A + (size_t)grow * IN_F + cs;
        v8h lo = *(const v8h*)(gp);
        v8h hi = *(const v8h*)(gp + 8);
        *(v8h*)(As + r * LDS_PITCH + cs)     = lo;
        *(v8h*)(As + r * LDS_PITCH + cs + 8) = hi;
    }
    __syncthreads();

    // --- per-wave: two 16x16 output tiles -----------------------------------
    const int tN0  = wave * 2;
    const int tN1  = wave * 2 + 1;
    const int ksub = (lane >> 4) << 3;             // lanes 16-31 shift K by 8
    const int arw  = (lane & 15) * LDS_PITCH + ksub;

    const _Float16* bp0 = W + (size_t)(tN0 * 16 + (lane & 15)) * IN_F + ksub;
    const _Float16* bp1 = W + (size_t)(tN1 * 16 + (lane & 15)) * IN_F + ksub;
    __builtin_prefetch(bp0, 0, 3);
    __builtin_prefetch(bp1, 0, 3);

    v8f acc0 = {};
    v8f acc1 = {};
    #pragma unroll
    for (int k0 = 0; k0 < IN_F; k0 += 32) {
        v16h a  = fuse16(*(const v8h*)(As + arw + k0),
                         *(const v8h*)(As + arw + k0 + 16));   // ds_load_b128 x2
        v16h b0 = load_half16(bp0 + k0);
        v16h b1 = load_half16(bp1 + k0);
        acc0 = __builtin_amdgcn_wmma_f32_16x16x32_f16(
            false, a, false, b0, (short)0, acc0, false, false);
        acc1 = __builtin_amdgcn_wmma_f32_16x16x32_f16(
            false, a, false, b1, (short)0, acc1, false, false);
    }

    // C/D layout: lanes 0-15 -> M rows 0..7, lanes 16-31 -> M rows 8..15
    const int mbase = tM * 16 + ((lane >> 4) << 3);
    #pragma unroll
    for (int tt = 0; tt < 2; ++tt) {
        const int ncol = (tt ? tN1 : tN0) * 16 + (lane & 15);
        const v8f& acc = tt ? acc1 : acc0;
        const float bb = bias[ncol];
        #pragma unroll
        for (int r = 0; r < 8; ++r) {
            const int mrow = mbase + r;
            if (mrow >= M) break;
            const float val = acc[r] + bb;
            if (MODE == 0) {
                outH[(size_t)mrow * ATT_D + ncol] = (_Float16)val;
            } else {
                const int pc = (ncol & (DK - 1)) * NHEAD + (ncol >> 5); // (dk,h)
                outF[(size_t)mrow * ATT_D + pc] = val;
            }
        }
    }
}

// --- edge logits + running max ---------------------------------------------

__global__ void __launch_bounds__(256)
edge_logits_max(const _Float16* __restrict__ qh, const _Float16* __restrict__ kh,
                const long long* __restrict__ edge, const float* __restrict__ ew,
                float* __restrict__ logits, float* __restrict__ mbuf, int E)
{
    const int t = blockIdx.x * blockDim.x + threadIdx.x;
    if (t >= E * NHEAD) return;
    const int e = t >> 3, h = t & 7;
    const long long s = edge[e];        // edge[0][e] : q index & softmax segment
    const long long d = edge[E + e];    // edge[1][e] : k index
    const _Float16* qp = qh + (size_t)s * ATT_D + h * DK;
    const _Float16* kp = kh + (size_t)d * ATT_D + h * DK;
    float acc = 0.0f;
    #pragma unroll
    for (int c = 0; c < 4; ++c) {
        v8h a = *(const v8h*)(qp + c * 8);
        v8h b = *(const v8h*)(kp + c * 8);
        #pragma unroll
        for (int i = 0; i < 8; ++i) acc += (float)a[i] * (float)b[i];
    }
    const float logit = acc * 0.17677669529663687f * ew[e];  // 1/sqrt(32)
    logits[t] = logit;
    atomicMaxFloat(mbuf + (size_t)s * NHEAD + h, logit);
}

// --- exp + segment sum ------------------------------------------------------

__global__ void __launch_bounds__(256)
edge_exp_sum(const long long* __restrict__ edge, float* __restrict__ logits,
             const float* __restrict__ mbuf, float* __restrict__ sbuf, int E)
{
    const int t = blockIdx.x * blockDim.x + threadIdx.x;
    if (t >= E * NHEAD) return;
    const int e = t >> 3, h = t & 7;
    const long long s = edge[e];
    const float ex = __expf(logits[t] - mbuf[(size_t)s * NHEAD + h]);
    logits[t] = ex;
    atomicAdd(sbuf + (size_t)s * NHEAD + h, ex);
}

// --- normalize --------------------------------------------------------------

__global__ void __launch_bounds__(256)
edge_normalize(const long long* __restrict__ edge, float* __restrict__ logits,
               const float* __restrict__ sbuf, int E)
{
    const int t = blockIdx.x * blockDim.x + threadIdx.x;
    if (t >= E * NHEAD) return;
    const int e = t >> 3, h = t & 7;
    const long long s = edge[e];
    logits[t] = logits[t] / (sbuf[(size_t)s * NHEAD + h] + 1e-16f);
}

// ---------------------------------------------------------------------------

extern "C" void kernel_launch(void* const* d_in, const int* in_sizes, int n_in,
                              void* d_out, int out_size, void* d_ws, size_t ws_size,
                              hipStream_t stream) {
    const float*     x    = (const float*)d_in[0];
    const long long* edge = (const long long*)d_in[1];
    const float*     ew   = (const float*)d_in[2];
    const float*     Wq   = (const float*)d_in[3];
    const float*     bq   = (const float*)d_in[4];
    const float*     Wk   = (const float*)d_in[5];
    const float*     bk   = (const float*)d_in[6];
    const float*     Wv   = (const float*)d_in[7];
    const float*     bv   = (const float*)d_in[8];

    const int N = in_sizes[0] / IN_F;
    const int E = in_sizes[2];

    // workspace layout (bytes)
    char* ws = (char*)d_ws;
    const size_t szXh = (size_t)N * IN_F * sizeof(_Float16);
    const size_t szW  = (size_t)ATT_D * IN_F * sizeof(_Float16);
    const size_t szQh = (size_t)N * ATT_D * sizeof(_Float16);
    _Float16* xh  = (_Float16*)(ws);
    _Float16* wqh = (_Float16*)(ws + szXh);
    _Float16* wkh = (_Float16*)(ws + szXh + szW);
    _Float16* wvh = (_Float16*)(ws + szXh + 2 * szW);
    _Float16* qh  = (_Float16*)(ws + szXh + 3 * szW);
    _Float16* kh  = (_Float16*)(ws + szXh + 3 * szW + szQh);
    float*    mb  = (float*)   (ws + szXh + 3 * szW + 2 * szQh);
    float*    sb  = (float*)   (ws + szXh + 3 * szW + 2 * szQh + (size_t)N * NHEAD * 4);

    float* att  = (float*)d_out;                     // [E, 8]
    float* vout = (float*)d_out + (size_t)E * NHEAD; // [N, 32, 8]

    // 1) precision conversion to f16
    {
        int n = N * IN_F;
        cvt_f32_to_f16<<<(n / 4 + 255) / 256, 256, 0, stream>>>(x, xh, n);
        int nw = ATT_D * IN_F;
        cvt_f32_to_f16<<<(nw / 4 + 255) / 256, 256, 0, stream>>>(Wq, wqh, nw);
        cvt_f32_to_f16<<<(nw / 4 + 255) / 256, 256, 0, stream>>>(Wk, wkh, nw);
        cvt_f32_to_f16<<<(nw / 4 + 255) / 256, 256, 0, stream>>>(Wv, wvh, nw);
    }

    // 2) init softmax max/sum buffers
    {
        int n = N * NHEAD;
        init_max_sum<<<(n + 255) / 256, 256, 0, stream>>>(mb, sb, n);
    }

    // 3) WMMA GEMMs: Q, K (f16 staging), V (fused permute to d_out)
    {
        const int Mtiles = (N + 15) >> 4;            // one block per 16-row slab
        gemm_qkv_wmma<0><<<Mtiles, 256, 0, stream>>>(xh, wqh, bq, qh,  nullptr, N);
        gemm_qkv_wmma<0><<<Mtiles, 256, 0, stream>>>(xh, wkh, bk, kh,  nullptr, N);
        gemm_qkv_wmma<1><<<Mtiles, 256, 0, stream>>>(xh, wvh, bv, nullptr, vout, N);
    }

    // 4) edge logits + scatter softmax
    {
        const int nt = E * NHEAD;
        const int blocks = (nt + 255) / 256;
        edge_logits_max<<<blocks, 256, 0, stream>>>(qh, kh, edge, ew, att, mb, E);
        edge_exp_sum   <<<blocks, 256, 0, stream>>>(edge, att, mb, sb, E);
        edge_normalize <<<blocks, 256, 0, stream>>>(edge, att, sb, E);
    }
}
